// LearnableDelayRNNBackbone_45088566673681
// MI455X (gfx1250) — matrix operand order repaired and checked by reading, not compiled, not run on hardware
//
#include <hip/hip_runtime.h>
#include <hip/hip_bf16.h>
#include <math.h>

// Problem sizes (fixed by the reference)
#define T_ 512
#define B_ 64
#define I_ 128
#define H_ 512
#define O_ 128
#define D_ 32
#define NBLK 32   // persistent grid: one block per 16 hidden columns

typedef __attribute__((ext_vector_type(16))) __bf16 v16bf;
typedef __attribute__((ext_vector_type(8)))  __bf16 v8bf;
typedef __attribute__((ext_vector_type(8)))  float  v8f;

static __device__ __forceinline__ v16bf cat8(v8bf lo, v8bf hi) {
  return __builtin_shufflevector(lo, hi, 0,1,2,3,4,5,6,7,8,9,10,11,12,13,14,15);
}

static __device__ __forceinline__ v8f wmma_bf16(v16bf a, v16bf b, v8f c) {
  // (neg_a, A, neg_b, B, c_mod, C, reuse_a, reuse_b)
  return __builtin_amdgcn_wmma_f32_16x16x32_bf16(false, a, false, b, (short)0, c, false, false);
}

// ---------------------------------------------------------------------------
// TDM: stage one timestep's x tile (64x128 bf16 = 16 KB) into LDS.
// D# packing per CDNA5 ISA ch.8.3/8.4: group0 = {count=1, lds_addr,
// global_addr[56:0], type=2}; group1 = {mask=0, data_size=1(2B), 1-row tile
// of B_*I_ elements}.  This toolchain exposes the 6-arg builtin form
// (clang-23 / therock-10.0): (u32x4, i32x8, i32x4, i32x4, i32x8, i32 cpol).
// ---------------------------------------------------------------------------
static __device__ __forceinline__ void tdm_load_x_tile(const __bf16* src,
                                                       unsigned lds_off) {
  typedef __attribute__((ext_vector_type(4))) unsigned int u32x4;
  typedef __attribute__((ext_vector_type(8))) int i32x8;
  typedef __attribute__((ext_vector_type(4))) int i32x4;
  const unsigned long long ga = (unsigned long long)(const void*)src;
  u32x4 g0;
  g0[0] = 1u;                                   // count=1 (valid user D#)
  g0[1] = lds_off;                              // LDS byte address of tile
  g0[2] = (unsigned)(ga & 0xffffffffull);       // global_addr[31:0]
  g0[3] = (unsigned)(ga >> 32) | (2u << 30);    // global_addr[56:32] | type=2
  const unsigned td0 = (unsigned)(B_ * I_);     // one contiguous line
  i32x8 g1;
  g1[0] = 1 << 16;                              // data_size=1 (2 bytes)
  g1[1] = (int)((td0 & 0xffffu) << 16);         // tensor_dim0[15:0]
  g1[2] = (int)((td0 >> 16) | (1u << 16));      // tensor_dim0[31:16]|dim1=1
  g1[3] = (int)(td0 << 16);                     // tile_dim0 = 8192
  g1[4] = 1;                                    // tile_dim1 = 1
  g1[5] = (int)td0;                             // tensor_dim0_stride
  g1[6] = 0;
  g1[7] = 0;
  i32x4 z4 = {0, 0, 0, 0};
  i32x8 z8 = {0, 0, 0, 0, 0, 0, 0, 0};
  __builtin_amdgcn_tensor_load_to_lds(g0, g1, z4, z4, z8, 0);
}

// ---------------------------------------------------------------------------
// Workspace zeroing (Wd scatter target, history, barrier counter must be 0)
// ---------------------------------------------------------------------------
__global__ void ldrnn_zero(uint4* __restrict__ p, size_t n16) {
  size_t i = (size_t)blockIdx.x * blockDim.x + threadIdx.x;
  const size_t stride = (size_t)gridDim.x * blockDim.x;
  uint4 z = make_uint4(0u, 0u, 0u, 0u);
  for (; i < n16; i += stride) p[i] = z;
}

// ---------------------------------------------------------------------------
// One-time credit-matrix argmax: forward value of the straight-through
// gumbel-softmax is the hard one-hot, so per (h,i) pick the winning delay and
// scatter lateral[h,i] into WMMA-B fragment order:
//   Wp[((kb*H + n)*2 + ((kin>>3)&1))*16 + ((kin>>4)<<3) + (kin&7)],  k=(d-1)*H+i, kb=k>>5
// ---------------------------------------------------------------------------
__global__ void ldrnn_build_w(const float* __restrict__ tau,
                              const float* __restrict__ sexp,
                              const float* __restrict__ u,
                              const float* __restrict__ lateral,
                              __bf16* __restrict__ Wp) {
  const int idx = blockIdx.x * blockDim.x + threadIdx.x;
  if (idx >= H_ * H_) return;
  const int n = idx >> 9;          // hidden_out
  const int i = idx & (H_ - 1);    // hidden_in
  const float tc = fminf(fmaxf(tau[idx], 1.0f), (float)D_);
  const float se = sexp[i];        // scale_exponent broadcasts over hidden_in
  int bestd = 0;
  float best = -INFINITY;
  for (int d = 0; d <= D_; ++d) {
    const float dist  = 1.0f + fabsf((float)d - tc);
    const float logit = __logf(__expf(-se * __logf(dist)) + 1e-8f);
    const float uu    = u[((size_t)d * H_ + n) * H_ + i];
    const float g     = -__logf(-__logf(uu + 1e-8f) + 1e-8f);
    const float pl    = logit + g;
    if (pl > best) { best = pl; bestd = d; }   // first-max tiebreak
  }
  if (bestd >= 1) {   // delay-0 slice dropped by cm[1:]
    const int k  = (bestd - 1) * H_ + i;
    const int kb = k >> 5, kin = k & 31;
    Wp[(((size_t)kb * H_ + n) * 2 + ((kin >> 3) & 1)) * 16 + ((kin >> 4) << 3) + (kin & 7)] =
        (__bf16)lateral[idx];
  }
}

// x -> bf16 (row-major, i contiguous)
__global__ void ldrnn_cvt_x(const float* __restrict__ x, __bf16* __restrict__ xb) {
  const int idx = blockIdx.x * blockDim.x + threadIdx.x;
  if (idx < T_ * B_ * I_) xb[idx] = (__bf16)x[idx];
}

// aff_w (H x I) packed as WMMA-B fragments over K=I
__global__ void ldrnn_pack_aff(const float* __restrict__ w, __bf16* __restrict__ Wa) {
  const int idx = blockIdx.x * blockDim.x + threadIdx.x;
  if (idx >= H_ * I_) return;
  const int n = idx / I_, k = idx % I_;
  const int kb = k >> 5, kin = k & 31;
  Wa[(((size_t)kb * H_ + n) * 2 + ((kin >> 3) & 1)) * 16 + ((kin >> 4) << 3) + (kin & 7)] = (__bf16)w[idx];
}

// eff_w (O x H) packed as WMMA-B fragments over K=H
__global__ void ldrnn_pack_eff(const float* __restrict__ w, __bf16* __restrict__ We) {
  const int idx = blockIdx.x * blockDim.x + threadIdx.x;
  if (idx >= O_ * H_) return;
  const int n = idx / H_, k = idx % H_;
  const int kb = k >> 5, kin = k & 31;
  We[(((size_t)kb * O_ + n) * 2 + ((kin >> 3) & 1)) * 16 + ((kin >> 4) << 3) + (kin & 7)] = (__bf16)w[idx];
}

// ---------------------------------------------------------------------------
// Monotonic grid barrier (counter never reset -> no reset race).
// ---------------------------------------------------------------------------
static __device__ __forceinline__ void grid_barrier(int* cnt, int seq) {
  __syncthreads();
  if (threadIdx.x == 0) {
    __threadfence();                 // release my writes device-wide
    atomicAdd(cnt, 1);
    volatile int* vc = cnt;
    while (*vc < NBLK * seq) __builtin_amdgcn_s_sleep(2);
    __threadfence();                 // acquire others' writes
  }
  __syncthreads();
}

// ---------------------------------------------------------------------------
// Persistent scan kernel: 32 blocks x 8 wave32.
// Phase 1: h_delayed (64 x 16 tile) = sum_{j=1..32} Wd[j] @ h_{t-j}
//          delay-outer / 16-i-block-inner (immediate-offset loads),
//          K split across wave pairs, LDS reduce.  TDM prefetches x_{t+1}.
// Phase 2a: h_new = tanh(x_t @ aff_w^T + aff_b + h_delayed) -> hist[p]
//           (x_t A-fragments read from the TDM-staged LDS tile)
// Phase 2b: y_t = h_delayed @ eff_w^T + eff_b  (blocks 0..7, waves 4..7)
// ---------------------------------------------------------------------------
__launch_bounds__(256, 1)
__global__ void ldrnn_scan(const __bf16* __restrict__ xb,
                           const __bf16* __restrict__ Wp,
                           const __bf16* __restrict__ Wa,
                           const __bf16* __restrict__ We,
                           __bf16* __restrict__ hist,   // (D, B, H) circular
                           float*  __restrict__ hD,     // (B, H) h_delayed
                           const float* __restrict__ aff_b,
                           const float* __restrict__ eff_b,
                           float* __restrict__ out,     // (T, B, O)
                           int* __restrict__ cnt) {
  const int wave = threadIdx.x >> 5;
  const int lane = threadIdx.x & 31;
  const int row  = lane & 15;   // A-fragment M row / B,C,D N column
  const int hi   = lane >> 4;   // K-half select per ISA fragment layout
  const int blk  = blockIdx.x;
  const int n0   = blk << 4;    // this block's 16 hidden columns

  // xlds first so its LDS offset is 0 (TDM lds_addr base); double-buffered.
  __shared__ __align__(128) __bf16 xlds[2][B_ * I_];   // 2 x 16 KB
  __shared__ float red[4][32][8];                      // K-split partial C

  // Preload x tile for t=0 (one wave issues; TENSORcnt tracked on that wave).
  if (wave == 0) tdm_load_x_tile(xb, 0u);

  int seq = 0;

  for (int t = 0; t < T_; ++t) {
    const int p = t & (D_ - 1);   // circular write pointer (== ptr in ref)

    // ---------------- Phase 1: delayed lateral contraction ----------------
    {
      const int mt = wave & 3;        // 16-batch m-tile
      const int kh = wave >> 2;       // delay half: j in [kh*16+1, kh*16+16]
      const int b  = (mt << 4) + row;
      const int j0 = (kh << 4) + 1;
      v8f acc0 = {}; v8f acc1 = {};   // two chains hide WMMA latency
      // B base for kb = (j0-1)*16; advances 512 KB per delay.
      const __bf16* bpl =
          Wp + ((((size_t)(j0 - 1) << 4) * H_ + (n0 + row)) * 2 + hi) * 16;
      #pragma unroll 1
      for (int jj = 0; jj < 16; ++jj) {
        const int j = j0 + jj;                  // delay
        const int s = (p - j) & (D_ - 1);       // history slot of h_{t-j}
        const __bf16* arl = hist + (size_t)(s * B_ + b) * H_ + hi * 8;
        __builtin_prefetch(bpl + (size_t)16 * H_ * 32, 0, 1);  // next delay's W
        #pragma unroll
        for (int ib = 0; ib < 16; ib += 2) {    // 16 i-blocks, imm offsets
          v16bf a0 = cat8(*(const v8bf*)(arl + (ib << 5)),
                          *(const v8bf*)(arl + (ib << 5) + 16));
          v16bf b0 = *(const v16bf*)(bpl + (size_t)ib * H_ * 32);
          acc0 = wmma_bf16(a0, b0, acc0);
          v16bf a1 = cat8(*(const v8bf*)(arl + ((ib + 1) << 5)),
                          *(const v8bf*)(arl + ((ib + 1) << 5) + 16));
          v16bf b1 = *(const v16bf*)(bpl + (size_t)(ib + 1) * H_ * 32);
          acc1 = wmma_bf16(a1, b1, acc1);
        }
        bpl += (size_t)16 * H_ * 32;
      }
      v8f acc = acc0 + acc1;
      if (kh == 1) {
        #pragma unroll
        for (int r = 0; r < 8; ++r) red[mt][lane][r] = acc[r];
      }
      __syncthreads();
      if (kh == 0) {
        #pragma unroll
        for (int r = 0; r < 8; ++r) {
          const int bb = (mt << 4) + (hi << 3) + r;   // C layout: M=hi*8+r
          hD[(size_t)bb * H_ + n0 + row] = acc[r] + red[mt][lane][r];
        }
      }
    }

    // Stage x tile for t+1 into the other LDS buffer (overlaps barrier).
    if (wave == 0) {
      const int tn = (t + 1 < T_) ? (t + 1) : (T_ - 1);
      tdm_load_x_tile(xb + (size_t)tn * B_ * I_,
                      (unsigned)((tn & 1) * (B_ * I_ * 2)));
      __builtin_amdgcn_s_wait_tensorcnt(0);  // current tile ready in LDS
    }
    grid_barrier(cnt, ++seq);   // hD complete + LDS tile published

    // ---------------- Phase 2a: h_new -> history[p] ----------------
    if (wave < 4) {
      const int mt = wave;
      const int b  = (mt << 4) + row;
      const __bf16* xrow = &xlds[t & 1][b * I_] + hi * 8;
      v8f acc = {};
      #pragma unroll
      for (int kb = 0; kb < 4; ++kb) {   // K = I = 128, LDS A-fragments
        v16bf a = cat8(*(const v8bf*)(xrow + (kb << 5)),
                       *(const v8bf*)(xrow + (kb << 5) + 16));
        v16bf bw =
            *(const v16bf*)(Wa + (((size_t)kb * H_ + (n0 + row)) * 2 + hi) * 16);
        acc = wmma_bf16(a, bw, acc);
      }
      const int nn = n0 + row;
      const float bias = aff_b[nn];
      #pragma unroll
      for (int r = 0; r < 8; ++r) {
        const int bb = (mt << 4) + (hi << 3) + r;
        const float v = acc[r] + hD[(size_t)bb * H_ + nn] + bias;
        hist[((size_t)p * B_ + bb) * H_ + nn] = (__bf16)tanhf(v);
      }
    } else if (blk < (O_ / 16)) {
      // -------------- Phase 2b: y_t = hD @ eff_w^T + eff_b --------------
      const int mt = wave - 4;
      const int o0 = blk << 4;
      const float* ar0 = hD + (size_t)((mt << 4) + row) * H_ + hi * 8;
      v8f acc = {};
      #pragma unroll 2
      for (int kb = 0; kb < 16; ++kb) {  // K = H = 512
        v8f flo = *(const v8f*)(ar0 + (kb << 5));
        v8f fhi = *(const v8f*)(ar0 + (kb << 5) + 16);
        v16bf a;
        #pragma unroll
        for (int q = 0; q < 8; ++q) {
          a[q]     = (__bf16)flo[q];
          a[q + 8] = (__bf16)fhi[q];
        }
        v16bf bw =
            *(const v16bf*)(We + (((size_t)kb * O_ + (o0 + row)) * 2 + hi) * 16);
        acc = wmma_bf16(a, bw, acc);
      }
      const int oo = o0 + row;
      const float bias = eff_b[oo];
      #pragma unroll
      for (int r = 0; r < 8; ++r) {
        const int bb = (mt << 4) + (hi << 3) + r;
        out[((size_t)t * B_ + bb) * O_ + oo] = acc[r] + bias;
      }
    }
    grid_barrier(cnt, ++seq);   // hist[p] + y visible before next step
  }
}

// ---------------------------------------------------------------------------
extern "C" void kernel_launch(void* const* d_in, const int* in_sizes, int n_in,
                              void* d_out, int out_size, void* d_ws, size_t ws_size,
                              hipStream_t stream) {
  (void)in_sizes; (void)n_in; (void)out_size; (void)ws_size;
  const float* x    = (const float*)d_in[0];
  const float* affw = (const float*)d_in[1];
  const float* affb = (const float*)d_in[2];
  const float* lat  = (const float*)d_in[3];
  const float* effw = (const float*)d_in[4];
  const float* effb = (const float*)d_in[5];
  const float* tau  = (const float*)d_in[6];
  const float* sexp = (const float*)d_in[7];
  const float* u    = (const float*)d_in[8];

  char* ws = (char*)d_ws;
  size_t off = 0;
  __bf16* Wp   = (__bf16*)(ws + off); off += (size_t)D_ * H_ * H_ * 2;  // 16 MB
  __bf16* Wa   = (__bf16*)(ws + off); off += (size_t)H_ * I_ * 2;
  __bf16* We   = (__bf16*)(ws + off); off += (size_t)O_ * H_ * 2;
  __bf16* xb   = (__bf16*)(ws + off); off += (size_t)T_ * B_ * I_ * 2;  // 8 MB
  __bf16* hist = (__bf16*)(ws + off); off += (size_t)D_ * B_ * H_ * 2;  // 2 MB
  float*  hD   = (float*)(ws + off);  off += (size_t)B_ * H_ * 4;
  int*    cnt  = (int*)(ws + off);    off += 256;

  // Zero scatter targets + history + barrier counter (deterministic per call).
  const size_t n16 = (off + 15) / 16;
  ldrnn_zero<<<4096, 256, 0, stream>>>((uint4*)ws, n16);

  ldrnn_build_w<<<(H_ * H_ + 255) / 256, 256, 0, stream>>>(tau, sexp, u, lat, Wp);
  ldrnn_cvt_x<<<(T_ * B_ * I_ + 255) / 256, 256, 0, stream>>>(x, xb);
  ldrnn_pack_aff<<<(H_ * I_ + 255) / 256, 256, 0, stream>>>(affw, Wa);
  ldrnn_pack_eff<<<(O_ * H_ + 255) / 256, 256, 0, stream>>>(effw, We);

  ldrnn_scan<<<NBLK, 256, 0, stream>>>(xb, Wp, Wa, We, hist, hD,
                                       affb, effb, (float*)d_out, cnt);
}
